// LSTMOnly_90108413870859
// MI455X (gfx1250) — matrix operand order repaired and checked by reading, not compiled, not run on hardware
//
#include <hip/hip_runtime.h>
#include <hip/hip_bf16.h>

#define HID   256
#define NG    1024
#define FUT   360
#define THIST 120
#define BT    32       // batch rows per workgroup
#define NTHR  512      // 16 waves: one hidden tile (jn) per wave

typedef __bf16 v16bf __attribute__((ext_vector_type(16)));
typedef float  v8f   __attribute__((ext_vector_type(8)));

__device__ __forceinline__ float sigmoidf_(float x) {
    return 1.0f / (1.0f + __expf(-x));
}
__device__ __forceinline__ float tanhf_(float x) {
    return 1.0f - 2.0f / (__expf(2.0f * x) + 1.0f);
}

// Load A fragment (16x32, contiguous k order, same as packed B):
// lanes 0-15  : row M=lane,    K = base+[0..16)
// lanes 16-31 : row M=lane-16, K = base+16+[0..16)
__device__ __forceinline__ v16bf ldA(const __bf16* rowp, int c0) {
    v16bf a;
    float4* ap = reinterpret_cast<float4*>(&a);
    ap[0] = *reinterpret_cast<const float4*>(rowp + c0);
    ap[1] = *reinterpret_cast<const float4*>(rowp + c0 + 8);
    return a;
}

// Load B fragment from pre-packed weights: 16 contiguous bf16 per lane (32B).
__device__ __forceinline__ v16bf ldB(const __bf16* __restrict__ p) {
    v16bf b;
    float4* bp = reinterpret_cast<float4*>(&b);
    bp[0] = *reinterpret_cast<const float4*>(p);
    bp[1] = *reinterpret_cast<const float4*>(p + 8);
    return b;
}

// Pack gate-weight matrix W (NG x Kdim, row-major fp32; optionally concat of
// two NG x 256 matrices along K) into per-fragment bf16 layout:
// out[(((ntile*KS + ks)*32 + lane)*16 + j] = W[ntile*16 + lane%16][ks*32 + (lane/16)*16 + j]
__global__ void lstm_pack_w(const float* __restrict__ W1,
                            const float* __restrict__ W2,
                            __bf16* __restrict__ out, int Kdim) {
    int idx = blockIdx.x * blockDim.x + threadIdx.x;
    int KS = Kdim / 32;
    int total = 64 * KS * 512;
    if (idx >= total) return;
    int j    = idx & 15;
    int lane = (idx >> 4) & 31;
    int rem  = idx >> 9;
    int ks    = rem % KS;
    int ntile = rem / KS;
    int n = ntile * 16 + (lane & 15);
    int k = ks * 32 + (lane >> 4) * 16 + j;
    float v;
    if (W2 == nullptr) v = W1[(size_t)n * Kdim + k];
    else               v = (k < 256) ? W1[(size_t)n * 256 + k]
                                     : W2[(size_t)n * 256 + (k - 256)];
    out[idx] = (__bf16)v;
}

// One LSTM layer for this wave's 2 logical tiles (2 M-tiles x 1 hidden tile),
// each logical tile = 4 gate accumulators (i,f,g,o) via WMMA, then lane-local
// gate nonlinearity + c update. New h stashed in registers (written after barrier).
__device__ __forceinline__ void layer_pass(
    const __bf16* __restrict__ P, int KS,
    const float* __restrict__ biasRow,
    const float (*xc)[2],            // xcur in LDS, or nullptr
    const float* __restrict__ Wx,    // layer-1 x-weights (NG x 2) raw, or nullptr
    const __bf16 (*hsrc)[520],       // hcat in LDS
    v8f* cst, __bf16 (*hstash)[8],
    int lane, int jn)
{
    const int lh = lane >> 4;
    const int nl = lane & 15;
    const size_t gstride = (size_t)16 * KS * 512;  // step between gate blocks
    const __bf16* bbase = P + ((size_t)jn * KS * 32 + lane) * 16;
    // per-lane gate biases (shared by both M-tiles)
    const float bi = biasRow[       jn * 16 + nl];
    const float bf = biasRow[256  + jn * 16 + nl];
    const float bg = biasRow[512  + jn * 16 + nl];
    const float bo = biasRow[768  + jn * 16 + nl];
    float2 w0 = make_float2(0.f, 0.f), w1 = w0, w2 = w0, w3 = w0;
    if (Wx) {   // rank-2 x-contribution weights, streamed from L2
        w0 = *reinterpret_cast<const float2*>(Wx + (size_t)(0 * 256 + jn * 16 + nl) * 2);
        w1 = *reinterpret_cast<const float2*>(Wx + (size_t)(1 * 256 + jn * 16 + nl) * 2);
        w2 = *reinterpret_cast<const float2*>(Wx + (size_t)(2 * 256 + jn * 16 + nl) * 2);
        w3 = *reinterpret_cast<const float2*>(Wx + (size_t)(3 * 256 + jn * 16 + nl) * 2);
    }
#pragma unroll 1   // tiles strictly sequential: one tile's accumulators live at a time
    for (int ii = 0; ii < 2; ++ii) {
        v8f acc0 = {}, acc1 = {}, acc2 = {}, acc3 = {};
        const __bf16* rowp = hsrc[ii * 16 + nl];
        // unroll-by-2: overlap next iteration's L2 loads with current WMMAs
        // without blowing the VGPR budget (full unroll spilled to scratch).
#pragma unroll 2
        for (int ks = 0; ks < KS; ++ks) {
            v16bf a = ldA(rowp, ks * 32 + lh * 16);
            const __bf16* bp = bbase + (size_t)ks * 512;
            v16bf b0 = ldB(bp);
            v16bf b1 = ldB(bp + gstride);
            v16bf b2 = ldB(bp + 2 * gstride);
            v16bf b3 = ldB(bp + 3 * gstride);
            acc0 = __builtin_amdgcn_wmma_f32_16x16x32_bf16(false, a, false, b0, (short)0, acc0, false, false);
            acc1 = __builtin_amdgcn_wmma_f32_16x16x32_bf16(false, a, false, b1, (short)0, acc1, false, false);
            acc2 = __builtin_amdgcn_wmma_f32_16x16x32_bf16(false, a, false, b2, (short)0, acc2, false, false);
            acc3 = __builtin_amdgcn_wmma_f32_16x16x32_bf16(false, a, false, b3, (short)0, acc3, false, false);
        }
        // lane-local elementwise: column n = g*256 + jn*16 + nl
#pragma unroll
        for (int v = 0; v < 8; ++v) {
            int row = ii * 16 + lh * 8 + v;   // C-layout: M = v (+8 for upper half-wave)
            float xa = 0.f, xb = 0.f;
            if (xc) { xa = xc[row][0]; xb = xc[row][1]; }
            float gi = sigmoidf_(acc0[v] + bi + xa * w0.x + xb * w0.y);
            float gf = sigmoidf_(acc1[v] + bf + xa * w1.x + xb * w1.y);
            float gg = tanhf_   (acc2[v] + bg + xa * w2.x + xb * w2.y);
            float go = sigmoidf_(acc3[v] + bo + xa * w3.x + xb * w3.y);
            float cn = gf * cst[ii][v] + gi * gg;
            cst[ii][v] = cn;
            hstash[ii][v] = (__bf16)(go * tanhf_(cn));
        }
    }
}

__device__ __forceinline__ void write_h(__bf16 (*hcat)[520], __bf16 (*hstash)[8],
                                        int colOff, int lane, int jn)
{
    const int lh = lane >> 4;
    const int nl = lane & 15;
#pragma unroll
    for (int ii = 0; ii < 2; ++ii)
#pragma unroll
        for (int v = 0; v < 8; ++v) {
            int row = ii * 16 + lh * 8 + v;
            hcat[row][colOff + jn * 16 + nl] = hstash[ii][v];
        }
}

__global__ __launch_bounds__(NTHR) void lstm_seq_kernel(
    const float* __restrict__ hist,
    const float* __restrict__ encWih0, const float* __restrict__ decWih0,
    const __bf16* __restrict__ Pe1, const __bf16* __restrict__ Pe2,
    const __bf16* __restrict__ Pd1, const __bf16* __restrict__ Pd2,
    const float* __restrict__ eb0a, const float* __restrict__ eb0b,
    const float* __restrict__ eb1a, const float* __restrict__ eb1b,
    const float* __restrict__ db0a, const float* __restrict__ db0b,
    const float* __restrict__ db1a, const float* __restrict__ db1b,
    const float* __restrict__ fcW, const float* __restrict__ fcb,
    float* __restrict__ out)
{
    __shared__ __align__(16) __bf16 hcat[BT][520];  // cols [0,256)=h1, [256,512)=h2
    __shared__ float biasl[4][NG];                  // combined bih+bhh per layer
    __shared__ float xcur[BT][2];

    const int tid   = threadIdx.x;
    const int lane  = tid & 31;
    const int jn    = tid >> 5;                     // 16 waves cover 16 hidden tiles
    const int bBase = blockIdx.x * BT;

    for (int i = tid; i < BT * 520; i += NTHR) ((__bf16*)hcat)[i] = (__bf16)0.0f;
    for (int i = tid; i < NG; i += NTHR) {
        biasl[0][i] = eb0a[i] + eb0b[i];
        biasl[1][i] = eb1a[i] + eb1b[i];
        biasl[2][i] = db0a[i] + db0b[i];
        biasl[3][i] = db1a[i] + db1b[i];
    }

    v8f c1[2], c2[2];
    __bf16 h1s[2][8], h2s[2][8];
#pragma unroll
    for (int t = 0; t < 2; ++t) { c1[t] = (v8f){}; c2[t] = (v8f){}; }
    __syncthreads();

    // ---------------- encoder ----------------
    for (int t = 0; t < THIST; ++t) {
        if (tid < BT * 2) {
            int r = tid >> 1, o = tid & 1;
            xcur[r][o] = hist[((size_t)(bBase + r) * THIST + t) * 2 + o];
        }
        __syncthreads();
        layer_pass(Pe1, 8,  biasl[0], xcur, encWih0, hcat, c1, h1s, lane, jn);
        __syncthreads();
        write_h(hcat, h1s, 0, lane, jn);
        __syncthreads();
        layer_pass(Pe2, 16, biasl[1], nullptr, nullptr, hcat, c2, h2s, lane, jn);
        __syncthreads();
        write_h(hcat, h2s, 256, lane, jn);
        __syncthreads();
    }

    // ---------------- decoder ---------------- (xcur == history[:, -1, :])
    for (int t = 0; t < FUT; ++t) {
        layer_pass(Pd1, 8,  biasl[2], xcur, decWih0, hcat, c1, h1s, lane, jn);
        __syncthreads();
        write_h(hcat, h1s, 0, lane, jn);
        __syncthreads();
        layer_pass(Pd2, 16, biasl[3], nullptr, nullptr, hcat, c2, h2s, lane, jn);
        __syncthreads();
        write_h(hcat, h2s, 256, lane, jn);
        __syncthreads();
        if (tid < BT * 2) {   // curr = curr + h2 @ fc_W.T + fc_b ; emit pred
            int r = tid >> 1, o = tid & 1;
            float acc = fcb[o];
            const float* w = fcW + o * HID;
            for (int k = 0; k < HID; ++k) acc += (float)hcat[r][256 + k] * w[k];
            float nv = xcur[r][o] + acc;
            xcur[r][o] = nv;
            out[(size_t)(bBase + r) * (FUT * 2) + t * 2 + o] = nv;
        }
        __syncthreads();
    }
}

extern "C" void kernel_launch(void* const* d_in, const int* in_sizes, int n_in,
                              void* d_out, int out_size, void* d_ws, size_t ws_size,
                              hipStream_t stream) {
    const float* hist     = (const float*)d_in[0];
    const float* encWih0  = (const float*)d_in[1];
    const float* encWhh0  = (const float*)d_in[2];
    const float* enc_bih0 = (const float*)d_in[3];
    const float* enc_bhh0 = (const float*)d_in[4];
    const float* encWih1  = (const float*)d_in[5];
    const float* encWhh1  = (const float*)d_in[6];
    const float* enc_bih1 = (const float*)d_in[7];
    const float* enc_bhh1 = (const float*)d_in[8];
    const float* decWih0  = (const float*)d_in[9];
    const float* decWhh0  = (const float*)d_in[10];
    const float* dec_bih0 = (const float*)d_in[11];
    const float* dec_bhh0 = (const float*)d_in[12];
    const float* decWih1  = (const float*)d_in[13];
    const float* decWhh1  = (const float*)d_in[14];
    const float* dec_bih1 = (const float*)d_in[15];
    const float* dec_bhh1 = (const float*)d_in[16];
    const float* fcW      = (const float*)d_in[17];
    const float* fcb      = (const float*)d_in[18];
    float* out = (float*)d_out;

    // Packed weight buffers in workspace (bf16): 512KB + 1MB + 512KB + 1MB = 3MB
    const int E1 = 64 * 8  * 512;   // K=256
    const int E2 = 64 * 16 * 512;   // K=512 (Wih1 ++ Whh1)
    __bf16* Pe1 = (__bf16*)d_ws;
    __bf16* Pe2 = Pe1 + E1;
    __bf16* Pd1 = Pe2 + E2;
    __bf16* Pd2 = Pd1 + E1;

    lstm_pack_w<<<(E1 + 255) / 256, 256, 0, stream>>>(encWhh0, nullptr, Pe1, 256);
    lstm_pack_w<<<(E2 + 255) / 256, 256, 0, stream>>>(encWih1, encWhh1, Pe2, 512);
    lstm_pack_w<<<(E1 + 255) / 256, 256, 0, stream>>>(decWhh0, nullptr, Pd1, 256);
    lstm_pack_w<<<(E2 + 255) / 256, 256, 0, stream>>>(decWih1, decWhh1, Pd2, 512);

    int B = in_sizes[0] / (THIST * 2);
    lstm_seq_kernel<<<B / BT, NTHR, 0, stream>>>(
        hist, encWih0, decWih0, Pe1, Pe2, Pd1, Pd2,
        enc_bih0, enc_bhh0, enc_bih1, enc_bhh1,
        dec_bih0, dec_bhh0, dec_bih1, dec_bhh1,
        fcW, fcb, out);
}
